// PINN_4964982194547
// MI455X (gfx1250) — matrix operand order repaired and checked by reading, not compiled, not run on hardware
//
#include <hip/hip_runtime.h>
#include <hip/hip_bf16.h>

// ---------------------------------------------------------------------------
// PINN forward + d/dt + d2/dt2 via forward-mode 2-jets, bf16 WMMA on gfx1250.
// One wave32 per 16-row tile. Per hidden layer the full 3-jet activation set
// (16 rows x 256 K) is loaded once into 192 VGPRs of A-fragments; the inner
// loop is then pure  global_load_b128(B) + 3x v_wmma_f32_16x16x32_bf16.
// Single in-place LDS buffer (25.5 KB) -> up to 12 waves per WGP.
// ---------------------------------------------------------------------------

typedef __attribute__((ext_vector_type(16))) __bf16 v16bf;
typedef __attribute__((ext_vector_type(8)))  float  v8f;

union ABf   { uint4 q[2]; v16bf v; };
union BfView{ uint4 q; unsigned short s[8]; };

#define STRIDE    544                 // 256 bf16 row + 32B pad (16B aligned)
#define JETBUF    (16 * STRIDE)       // 8704 B : one jet, 16 rows
#define LDS_BYTES (3 * JETBUF)        // 26112 B : {value,d1,d2}, single buffer

__device__ __forceinline__ unsigned short f2bf(float x) {
    union { float f; unsigned u; } c; c.f = x;
    unsigned r = c.u + 0x7FFFu + ((c.u >> 16) & 1u);   // round-to-nearest-even
    return (unsigned short)(r >> 16);
}
__device__ __forceinline__ float bf2f(unsigned short b) {
    union { unsigned u; float f; } c; c.u = ((unsigned)b) << 16;
    return c.f;
}
__device__ __forceinline__ float tanh_fast(float x) {
    float ax = fabsf(x);
    float e  = __expf(-2.f * ax);                      // v_exp_f32 path
    float r  = __fdividef(1.f - e, 1.f + e);
    return copysignf(r, x);
}

// ---------------------------------------------------------------------------
// Prep: pack W1,W2,W3 (256x256 f32, row-major, W[k][n]) into bf16 WMMA
// B-fragment chunks. Chunk(l, ntile, ktile) = 512 bf16 (1KB):
//   lane = ((klocal>>3)&1 ? 16+nlocal : nlocal), pos = (klocal&7) + 8*(klocal>>4)
//   chunk[lane*16 + pos] = W[k][n]
// Main kernel loads a B fragment as two b128 loads at chunk + lane*32.
// ---------------------------------------------------------------------------
__global__ void pinn_pack_w(const float* __restrict__ W1,
                            const float* __restrict__ W2,
                            const float* __restrict__ W3,
                            unsigned short* __restrict__ ws) {
    int tid = blockIdx.x * blockDim.x + threadIdx.x;
    if (tid >= 3 * 65536) return;
    int l   = tid >> 16;
    int rem = tid & 65535;
    int k = rem >> 8;        // fan-in index
    int n = rem & 255;       // fan-out index
    const float* W = (l == 0) ? W1 : (l == 1) ? W2 : W3;
    float v = W[k * 256 + n];
    int kt = k >> 5, kl = k & 31;
    int nt = n >> 4, nl = n & 15;
    int lane = ((kl >> 3) & 1) ? (16 + nl) : nl;
    int pos  = (kl & 7) + ((kl >> 4) << 3);
    ws[l * 65536 + (nt * 8 + kt) * 512 + lane * 16 + pos] = f2bf(v);
}

// ---------------------------------------------------------------------------
// Main: one wave (32 threads) per 16 samples.
// ---------------------------------------------------------------------------
__global__ void __launch_bounds__(32)
pinn_main(const float* __restrict__ p,  const float* __restrict__ t,
          const float* __restrict__ W0, const float* __restrict__ b0,
          const float* __restrict__ b1, const float* __restrict__ b2,
          const float* __restrict__ b3,
          const float* __restrict__ W4, const float* __restrict__ b4,
          const float* __restrict__ sm, const float* __restrict__ sd,
          const float* __restrict__ sB,
          const uint4* __restrict__ wq,
          float* __restrict__ out, int N)
{
    __shared__ __attribute__((aligned(16))) char smem[LDS_BYTES];

    const int lane = threadIdx.x;     // 0..31
    const int lr   = lane & 15;       // row within tile / col within N-tile
    const int hi   = lane >> 4;       // lane half
    const int rowg = blockIdx.x * 16 + lr;
    const int rowc = (rowg < N) ? rowg : (N - 1);

    const float pv = p[rowc];
    const float tv = t[rowc];

    // ---- layer 0: [p,t] -> 256, jet seed dx/dt = (0,1) ----------------------
    {
        const int cbase = hi * 128;                 // lane half splits columns
        const size_t roff = (size_t)lr * STRIDE;
        #pragma unroll 4
        for (int j = 0; j < 128; ++j) {
            int c = cbase + j;
            float w0 = W0[c];            // W0[0][c]
            float w1 = W0[256 + c];      // W0[1][c] == dz/dt
            float z  = fmaf(pv, w0, fmaf(tv, w1, b0[c]));
            float h  = tanh_fast(z);
            float s  = 1.f - h * h;
            float h1 = s * w1;                      // s * z1
            float h2 = -2.f * h * h1 * w1;          // s * (z2 - 2 h z1^2), z2=0
            size_t off = roff + (size_t)(c << 1);
            *(unsigned short*)(smem + 0 * JETBUF + off) = f2bf(h);
            *(unsigned short*)(smem + 1 * JETBUF + off) = f2bf(h1);
            *(unsigned short*)(smem + 2 * JETBUF + off) = f2bf(h2);
        }
    }

    // ---- layers 1..3: 256 -> 256 --------------------------------------------
    // Entire 3-jet layer input held in 192 VGPRs of A-fragments; inner loop is
    // B-fragment global loads + 3 WMMAs sharing B. LDS buffer overwritten
    // in place by the epilogue (A already register-resident).
    for (int l = 0; l < 3; ++l) {
        const float* bias = (l == 0) ? b1 : (l == 1) ? b2 : b3;
        const uint4* wl = wq + (size_t)l * 8192;    // 128 chunks * 64 uint4

        // Load all A fragments for this layer (dense 16-bit 16x32 layout):
        // lanes 0-15 : K 0..7  @ +0,  K 16..23 @ +32
        // lanes16-31 : K 8..15 @ +16, K 24..31 @ +48
        ABf A0[8], A1[8], A2[8];
        #pragma unroll
        for (int kt = 0; kt < 8; ++kt) {
            size_t aoff = (size_t)lr * STRIDE + (size_t)(kt * 64 + hi * 16);
            const char* pz = smem + 0 * JETBUF + aoff;
            const char* p1 = smem + 1 * JETBUF + aoff;
            const char* p2 = smem + 2 * JETBUF + aoff;
            A0[kt].q[0] = *(const uint4*)pz; A0[kt].q[1] = *(const uint4*)(pz + 32);
            A1[kt].q[0] = *(const uint4*)p1; A1[kt].q[1] = *(const uint4*)(p1 + 32);
            A2[kt].q[0] = *(const uint4*)p2; A2[kt].q[1] = *(const uint4*)(p2 + 32);
        }

        for (int nt = 0; nt < 16; ++nt) {
            float bz = bias[nt * 16 + lr];
            v8f az = {}, a1 = {}, a2 = {};
            const uint4* wn = wl + nt * 512 + lane * 2;
            __builtin_prefetch(wl + (nt + 1) * 512 + lane * 2, 0, 3);

            #pragma unroll
            for (int kt = 0; kt < 8; ++kt) {
                ABf bfr;                              // 32 bytes/lane of W tile
                bfr.q[0] = wn[kt * 64];
                bfr.q[1] = wn[kt * 64 + 1];
                az = __builtin_amdgcn_wmma_f32_16x16x32_bf16(false, A0[kt].v, false, bfr.v, (short)0, az, false, false);
                a1 = __builtin_amdgcn_wmma_f32_16x16x32_bf16(false, A1[kt].v, false, bfr.v, (short)0, a1, false, false);
                a2 = __builtin_amdgcn_wmma_f32_16x16x32_bf16(false, A2[kt].v, false, bfr.v, (short)0, a2, false, false);
            }

            // epilogue: bias + tanh jet, write next activations in place
            const int col = nt * 16 + lr;           // D layout: lane -> column
            #pragma unroll
            for (int i = 0; i < 8; ++i) {
                int M = i + hi * 8;                 // D layout: vgpr+half -> row
                float z  = az[i] + bz;
                float z1 = a1[i];
                float z2 = a2[i];
                float h  = tanh_fast(z);
                float s  = 1.f - h * h;
                float h1 = s * z1;
                float h2 = s * (z2 - 2.f * h * z1 * z1);
                size_t off = (size_t)M * STRIDE + (size_t)(col << 1);
                *(unsigned short*)(smem + 0 * JETBUF + off) = f2bf(h);
                *(unsigned short*)(smem + 1 * JETBUF + off) = f2bf(h1);
                *(unsigned short*)(smem + 2 * JETBUF + off) = f2bf(h2);
            }
        }
    }

    // ---- layer 4: 256 -> 1, then PDE residual -------------------------------
    {
        float su = 0.f, s1 = 0.f, s2 = 0.f;
        const int cb = hi * 128;                    // split reduction per half
        const size_t roff = (size_t)lr * STRIDE + (size_t)(cb << 1);
        #pragma unroll 2
        for (int ch = 0; ch < 16; ++ch) {
            BfView qz, q1, q2;
            qz.q = *(const uint4*)(smem + 0 * JETBUF + roff + ch * 16);
            q1.q = *(const uint4*)(smem + 1 * JETBUF + roff + ch * 16);
            q2.q = *(const uint4*)(smem + 2 * JETBUF + roff + ch * 16);
            #pragma unroll
            for (int e = 0; e < 8; ++e) {
                float w = W4[cb + ch * 8 + e];
                su = fmaf(bf2f(qz.s[e]), w, su);
                s1 = fmaf(bf2f(q1.s[e]), w, s1);
                s2 = fmaf(bf2f(q2.s[e]), w, s2);
            }
        }
        su += __shfl_xor(su, 16, 32);
        s1 += __shfl_xor(s1, 16, 32);
        s2 += __shfl_xor(s2, 16, 32);

        if (hi == 0 && rowg < N) {
            float u = su + b4[0];
            float f = fmaf(sm[0], s2, fmaf(sd[0], s1, fmaf(sB[0], __sinf(u), -pv)));
            out[rowg]     = u;   // u_hat
            out[N + rowg] = f;   // residual
        }
    }
}

extern "C" void kernel_launch(void* const* d_in, const int* in_sizes, int n_in,
                              void* d_out, int out_size, void* d_ws, size_t ws_size,
                              hipStream_t stream) {
    const float* p  = (const float*)d_in[0];
    const float* t  = (const float*)d_in[1];
    const float* W0 = (const float*)d_in[2];
    const float* b0 = (const float*)d_in[3];
    const float* W1 = (const float*)d_in[4];
    const float* b1 = (const float*)d_in[5];
    const float* W2 = (const float*)d_in[6];
    const float* b2 = (const float*)d_in[7];
    const float* W3 = (const float*)d_in[8];
    const float* b3 = (const float*)d_in[9];
    const float* W4 = (const float*)d_in[10];
    const float* b4 = (const float*)d_in[11];
    const float* m  = (const float*)d_in[12];
    const float* d  = (const float*)d_in[13];
    const float* B  = (const float*)d_in[14];
    const int N = in_sizes[0];

    // pack hidden-layer weights into bf16 WMMA B-fragment chunks (384 KB)
    pinn_pack_w<<<768, 256, 0, stream>>>(W1, W2, W3, (unsigned short*)d_ws);

    // 16 samples per wave32 workgroup
    pinn_main<<<(N + 15) / 16, 32, 0, stream>>>(p, t, W0, b0, b1, b2, b3, W4, b4,
                                                m, d, B, (const uint4*)d_ws,
                                                (float*)d_out, N);
}
